// MDDformerImproved_7722351198543
// MI455X (gfx1250) — matrix operand order repaired and checked by reading, not compiled, not run on hardware
//
#include <hip/hip_runtime.h>
#include <math.h>

#define B_  8
#define T_  1024
#define D_  128
#define H_  8
#define DH_ 16
#define F_  256
#define L_  2
#define BT_ (B_*T_)

typedef __bf16 bf16;
typedef __attribute__((ext_vector_type(16))) __bf16 v16bf;
typedef __attribute__((ext_vector_type(8)))  __bf16 v8bf;
typedef __attribute__((ext_vector_type(8)))  float  v8f;
typedef __attribute__((ext_vector_type(4)))  unsigned int u32x4;
typedef __attribute__((ext_vector_type(8)))  int i32x8;
typedef __attribute__((ext_vector_type(4)))  int i32x4;

__device__ __forceinline__ v8bf ld8(const bf16* p) {
  return *reinterpret_cast<const v8bf*>(p);
}
__device__ __forceinline__ v16bf pack16(v8bf lo, v8bf hi) {
  v16bf r;
  #pragma unroll
  for (int i = 0; i < 8; ++i) { r[i] = lo[i]; r[i + 8] = hi[i]; }
  return r;
}

// --------------------------------------------------------------- TDM (2-D tile)
// Build D# per CDNA5 ISA and issue TENSOR_LOAD_TO_LDS.
// Tile: rows x width bf16, global row stride = stride_elems; LDS receives the
// tile packed contiguously (row stride = width elements).
// This toolchain declares the 6-arg builtin:
//   (uint32x4 g0, int32x8 g1, int32x4 g2, int32x4 g3, int32x8 extra, i32 cpol)
__device__ __forceinline__ void tdm_load_tile_2d(const bf16* gsrc,
                                                 unsigned lds_byte_off,
                                                 unsigned width, unsigned rows,
                                                 unsigned stride_elems) {
  const unsigned long long ga = (unsigned long long)(size_t)gsrc;
  u32x4 g0;
  g0[0] = 1u;                                            // count=1, user mode
  g0[1] = lds_byte_off;                                  // lds_addr
  g0[2] = (unsigned)(ga & 0xffffffffu);                  // global_addr[31:0]
  g0[3] = (unsigned)((ga >> 32) & 0x1ffffffu) | (2u << 30); // addr[56:32] | type=2
  i32x8 g1;
  g1[0] = (int)(1u << 16);                    // data_size code 1 (2 bytes)
  g1[1] = (int)(width << 16);                 // tensor_dim0[15:0] @ bits 63:48
  g1[2] = (int)((width >> 16) | (rows << 16));// dim0[31:16] | tensor_dim1[15:0]
  g1[3] = (int)((rows >> 16) | (width << 16));// dim1[31:16] | tile_dim0
  g1[4] = (int)rows;                          // tile_dim1 (tile_dim2 = 0)
  g1[5] = (int)stride_elems;                  // tensor_dim0_stride[31:0]
  g1[6] = 0;                                  // stride0[47:32] | dim1_stride lo
  g1[7] = 0;
  const i32x4 gz4 = {0, 0, 0, 0};             // groups 2/3 unused (2-D tensor)
  const i32x8 gz8 = {0, 0, 0, 0, 0, 0, 0, 0}; // extra descriptor words unused
  __builtin_amdgcn_tensor_load_to_lds(g0, g1, gz4, gz4, gz8, 0);
}

// ---------------------------------------------------------------- fp32 -> bf16
__global__ void cvt_bf16_kernel(const float* __restrict__ src,
                                bf16* __restrict__ dst, int n) {
  int i = blockIdx.x * 256 + threadIdx.x;
  if (i < n) dst[i] = (bf16)src[i];
}

// ---------------------------------------------------------------- LayerNorm
__global__ __launch_bounds__(128) void ln_kernel(const float* __restrict__ X,
                                                 const float* __restrict__ g,
                                                 const float* __restrict__ bsh,
                                                 bf16* __restrict__ out) {
  __shared__ float s1[128];
  __shared__ float s2[128];
  const int row = blockIdx.x, tid = threadIdx.x;
  const float v = X[(size_t)row * D_ + tid];
  s1[tid] = v; s2[tid] = v * v;
  __syncthreads();
  for (int st = 64; st > 0; st >>= 1) {
    if (tid < st) { s1[tid] += s1[tid + st]; s2[tid] += s2[tid + st]; }
    __syncthreads();
  }
  const float mean = s1[0] * (1.0f / D_);
  const float var  = s2[0] * (1.0f / D_) - mean * mean;
  const float o = (v - mean) * rsqrtf(var + 1e-5f) * g[tid] + bsh[tid];
  out[(size_t)row * D_ + tid] = (bf16)o;
}

// ---------------------------------------------------------------- GEMM (WMMA)
// C[M,N] = act(A[M,K](bf16) * W[K,N](bf16) + bias) (+resid); out fp32/bf16.
// Block = 8 waves; wave tile 16(M) x 64(N); block tile 64(M) x 128(N).
// W k-slabs (32x128) are DMA'd into LDS by the Tensor Data Mover, double
// buffered and pipelined on TENSORcnt; A streams from global with prefetch.
__global__ __launch_bounds__(256) void gemm_bf16_wmma(
    const bf16* __restrict__ A, const bf16* __restrict__ W,
    const float* __restrict__ bias, const float* __restrict__ resid,
    float* __restrict__ outF, bf16* __restrict__ outB,
    int M, int K, int N, int dogelu) {
  __shared__ bf16 Ws[2][32][128];          // double-buffered W slab (16 KB)

  const int t     = threadIdx.x;
  const int wave  = t >> 5;
  const int lane  = t & 31;
  const int lrow  = lane & 15;
  const int khalf = lane >> 4;
  const int mrow  = blockIdx.x * 64 + (wave >> 1) * 16;
  const int nblk  = blockIdx.y * 128;
  const int n0    = (wave & 1) * 64;

  v8f acc[4];
  #pragma unroll
  for (int j = 0; j < 4; ++j) acc[j] = (v8f){0.f,0.f,0.f,0.f,0.f,0.f,0.f,0.f};

  const bf16* arow_p = A + (size_t)(mrow + lrow) * K;
  const int   bk     = lrow + 16 * khalf;    // B-fragment K row for this lane
  const int   nk     = K >> 5;               // number of 32-wide K slabs

  const unsigned lds0 = (unsigned)(size_t)&Ws[0][0][0];
  const unsigned lds1 = (unsigned)(size_t)&Ws[1][0][0];

  if (wave == 0)                              // prime the pipeline: slab 0
    tdm_load_tile_2d(W + nblk, lds0, 128, 32, (unsigned)N);

  for (int kt = 0; kt < nk; ++kt) {
    const int k0 = kt << 5;
    __syncthreads();                          // all readers done with buf (kt+1)&1
    if (wave == 0) {
      if (kt + 1 < nk) {                      // DMA next slab, keep 1 in flight
        tdm_load_tile_2d(W + (size_t)(k0 + 32) * N + nblk,
                         (kt & 1) ? lds0 : lds1, 128, 32, (unsigned)N);
        __builtin_amdgcn_s_wait_tensorcnt(1); // slab kt resident (in-order)
      } else {
        __builtin_amdgcn_s_wait_tensorcnt(0);
      }
    }
    __syncthreads();                          // DMA data visible to all waves

    if (kt + 1 < nk) __builtin_prefetch(arow_p + k0 + 32, 0, 1);
    // A fragment: lane = row M, e0..7 = K[khalf*8..], e8..15 = K[16+khalf*8..]
    const v16bf afrag = pack16(ld8(arow_p + k0 + khalf * 8),
                               ld8(arow_p + k0 + 16 + khalf * 8));
    // B fragments from LDS: lane = N column, elements = K rows in table order
    const bf16 (*wsb)[128] = Ws[kt & 1];
    v16bf bfr[4];
    #pragma unroll
    for (int j = 0; j < 4; ++j)
      #pragma unroll
      for (int s = 0; s < 16; ++s)
        bfr[j][s] = wsb[(s < 8 ? 8 * khalf + s : 16 + 8 * khalf + (s - 8))][n0 + j * 16 + lrow];
    #pragma unroll
    for (int j = 0; j < 4; ++j)
      acc[j] = __builtin_amdgcn_wmma_f32_16x16x32_bf16(
          false, afrag, false, bfr[j], (short)0, acc[j], false, false);
  }

  #pragma unroll
  for (int j = 0; j < 4; ++j) {
    const int col = nblk + n0 + j * 16 + lrow;
    const float bv = bias ? bias[col] : 0.0f;
    #pragma unroll
    for (int r = 0; r < 8; ++r) {
      const int row = mrow + r + 8 * khalf;
      float v = acc[j][r] + bv;
      if (dogelu) v = 0.5f * v * (1.0f + erff(v * 0.70710678118654752f));
      const size_t idx = (size_t)row * N + col;
      if (resid) v += resid[idx];
      if (outF) outF[idx] = v;
      if (outB) outB[idx] = (bf16)v;
    }
  }
}

// ---------------------------------------------------------------- Attention
// One wave per (b, h, 16-query tile). Streaming softmax over 32-key tiles.
// S = [Qx|Qy](16x32) . [Kx;Ky](32x16) in ONE bf16 WMMA (DH=16 concat -> K=32).
__global__ __launch_bounds__(32) void attn_wmma(
    const bf16* __restrict__ qx, const bf16* __restrict__ kx, const bf16* __restrict__ vx,
    const bf16* __restrict__ qy, const bf16* __restrict__ ky, const bf16* __restrict__ vy,
    bf16* __restrict__ ox, bf16* __restrict__ oy) {
  __shared__ bf16 KTx[16][32];   // [dim][key] transposed K-tile, x stream
  __shared__ bf16 KTy[16][32];   // [dim][key] transposed K-tile, y stream
  __shared__ bf16 Ps [16][32];   // [query][key] probabilities (bf16)

  const int lane  = threadIdx.x;
  const int lrow  = lane & 15;
  const int khalf = lane >> 4;
  const int q0 = blockIdx.x * 16;
  const int h  = blockIdx.y;
  const int b  = blockIdx.z;
  const size_t hb = (size_t)h * DH_;

  const size_t qbase = ((size_t)(b * T_ + q0 + lrow) * D_) + hb + khalf * 8;
  const v16bf qfrag = pack16(ld8(qx + qbase), ld8(qy + qbase));

  float m[8], l[8];
  #pragma unroll
  for (int r = 0; r < 8; ++r) { m[r] = -3.0e38f; l[r] = 0.0f; }
  v8f accx = (v8f){0.f,0.f,0.f,0.f,0.f,0.f,0.f,0.f};
  v8f accy = (v8f){0.f,0.f,0.f,0.f,0.f,0.f,0.f,0.f};

  for (int s0 = 0; s0 < T_; s0 += 32) {
    const size_t krow = ((size_t)(b * T_ + s0 + lane) * D_) + hb;
    const v8bf kxa = ld8(kx + krow), kxb = ld8(kx + krow + 8);
    const v8bf kya = ld8(ky + krow), kyb = ld8(ky + krow + 8);
    __syncthreads();                       // previous iteration's LDS reads done
    #pragma unroll
    for (int d = 0; d < 8; ++d) {
      KTx[d][lane] = kxa[d]; KTx[d + 8][lane] = kxb[d];
      KTy[d][lane] = kya[d]; KTy[d + 8][lane] = kyb[d];
    }
    __syncthreads();

    v16bf kf0, kf1;
    #pragma unroll
    for (int s = 0; s < 8; ++s) {
      kf0[s] = KTx[8 * khalf + s][lrow];      kf0[s + 8] = KTy[8 * khalf + s][lrow];
      kf1[s] = KTx[8 * khalf + s][16 + lrow]; kf1[s + 8] = KTy[8 * khalf + s][16 + lrow];
    }
    const v8f zero = (v8f){0.f,0.f,0.f,0.f,0.f,0.f,0.f,0.f};
    const v8f sv0 = __builtin_amdgcn_wmma_f32_16x16x32_bf16(
        false, qfrag, false, kf0, (short)0, zero, false, false);
    const v8f sv1 = __builtin_amdgcn_wmma_f32_16x16x32_bf16(
        false, qfrag, false, kf1, (short)0, zero, false, false);

    #pragma unroll
    for (int r = 0; r < 8; ++r) {
      const float a = sv0[r] * 0.125f;     // 0.5 / sqrt(DH)
      const float c = sv1[r] * 0.125f;
      float mx = fmaxf(a, c);
      mx = fmaxf(mx, __shfl_xor(mx, 1, 32));
      mx = fmaxf(mx, __shfl_xor(mx, 2, 32));
      mx = fmaxf(mx, __shfl_xor(mx, 4, 32));
      mx = fmaxf(mx, __shfl_xor(mx, 8, 32));
      const float mn = fmaxf(m[r], mx);
      const float al = expf(m[r] - mn);
      const float p0 = expf(a - mn);
      const float p1 = expf(c - mn);
      float rs = p0 + p1;
      rs += __shfl_xor(rs, 1, 32);
      rs += __shfl_xor(rs, 2, 32);
      rs += __shfl_xor(rs, 4, 32);
      rs += __shfl_xor(rs, 8, 32);
      l[r] = l[r] * al + rs;
      m[r] = mn;
      accx[r] *= al; accy[r] *= al;
      Ps[r + 8 * khalf][lrow]      = (bf16)p0;
      Ps[r + 8 * khalf][16 + lrow] = (bf16)p1;
    }
    __syncthreads();

    const bf16* pr = &Ps[lrow][0];
    const v16bf pfrag = pack16(*(const v8bf*)(pr + khalf * 8),
                               *(const v8bf*)(pr + 16 + khalf * 8));
    v16bf vxf, vyf;
    #pragma unroll
    for (int s = 0; s < 16; ++s) {
      const int key = s < 8 ? 8 * khalf + s : 16 + 8 * khalf + (s - 8);
      const size_t vi = ((size_t)(b * T_ + s0 + key) * D_) + hb + lrow;
      vxf[s] = vx[vi]; vyf[s] = vy[vi];
    }
    accx = __builtin_amdgcn_wmma_f32_16x16x32_bf16(
        false, pfrag, false, vxf, (short)0, accx, false, false);
    accy = __builtin_amdgcn_wmma_f32_16x16x32_bf16(
        false, pfrag, false, vyf, (short)0, accy, false, false);
  }

  #pragma unroll
  for (int r = 0; r < 8; ++r) {
    const float inv = 1.0f / l[r];
    const size_t o = ((size_t)(b * T_ + q0 + r + 8 * khalf) * D_) + hb + lrow;
    ox[o] = (bf16)(accx[r] * inv);
    oy[o] = (bf16)(accy[r] * inv);
  }
}

// ---------------------------------------------------------------- host driver
extern "C" void kernel_launch(void* const* d_in, const int* in_sizes, int n_in,
                              void* d_out, int out_size, void* d_ws, size_t ws_size,
                              hipStream_t stream) {
  (void)in_sizes; (void)n_in; (void)out_size; (void)ws_size;
  const float* x    = (const float*)d_in[0];
  const float* y    = (const float*)d_in[1];
  const float* Wq   = (const float*)d_in[2];
  const float* Wk   = (const float*)d_in[3];
  const float* Wv   = (const float*)d_in[4];
  const float* Wox  = (const float*)d_in[5];
  const float* box  = (const float*)d_in[6];
  const float* Woy  = (const float*)d_in[7];
  const float* boy  = (const float*)d_in[8];
  const float* ln1xg = (const float*)d_in[9];
  const float* ln1xb = (const float*)d_in[10];
  const float* ln1yg = (const float*)d_in[11];
  const float* ln1yb = (const float*)d_in[12];
  const float* ln2xg = (const float*)d_in[13];
  const float* ln2xb = (const float*)d_in[14];
  const float* ln2yg = (const float*)d_in[15];
  const float* ln2yb = (const float*)d_in[16];
  const float* fxw1 = (const float*)d_in[17];
  const float* fxb1 = (const float*)d_in[18];
  const float* fxw2 = (const float*)d_in[19];
  const float* fxb2 = (const float*)d_in[20];
  const float* fyw1 = (const float*)d_in[21];
  const float* fyb1 = (const float*)d_in[22];
  const float* fyw2 = (const float*)d_in[23];
  const float* fyb2 = (const float*)d_in[24];

  char* wsp = (char*)d_ws;
  size_t off = 0;
  auto alloc = [&](size_t bytes) -> char* {
    char* p = wsp + off;
    off = (off + bytes + 255) & ~(size_t)255;
    return p;
  };

  bf16* Wqb  = (bf16*)alloc((size_t)L_ * D_ * D_ * 2);
  bf16* Wkb  = (bf16*)alloc((size_t)L_ * D_ * D_ * 2);
  bf16* Wvb  = (bf16*)alloc((size_t)L_ * D_ * D_ * 2);
  bf16* Woxb = (bf16*)alloc((size_t)L_ * D_ * D_ * 2);
  bf16* Woyb = (bf16*)alloc((size_t)L_ * D_ * D_ * 2);
  bf16* w1xb = (bf16*)alloc((size_t)L_ * D_ * F_ * 2);
  bf16* w2xb = (bf16*)alloc((size_t)L_ * F_ * D_ * 2);
  bf16* w1yb = (bf16*)alloc((size_t)L_ * D_ * F_ * 2);
  bf16* w2yb = (bf16*)alloc((size_t)L_ * F_ * D_ * 2);
  bf16* xn  = (bf16*)alloc((size_t)BT_ * D_ * 2);
  bf16* yn  = (bf16*)alloc((size_t)BT_ * D_ * 2);
  bf16* qxb = (bf16*)alloc((size_t)BT_ * D_ * 2);
  bf16* kxb = (bf16*)alloc((size_t)BT_ * D_ * 2);
  bf16* vxb = (bf16*)alloc((size_t)BT_ * D_ * 2);
  bf16* qyb = (bf16*)alloc((size_t)BT_ * D_ * 2);
  bf16* kyb = (bf16*)alloc((size_t)BT_ * D_ * 2);
  bf16* vyb = (bf16*)alloc((size_t)BT_ * D_ * 2);
  bf16* oxb = (bf16*)alloc((size_t)BT_ * D_ * 2);
  bf16* oyb = (bf16*)alloc((size_t)BT_ * D_ * 2);
  bf16* hx  = (bf16*)alloc((size_t)BT_ * F_ * 2);
  bf16* hy  = (bf16*)alloc((size_t)BT_ * F_ * 2);
  float* xcur = (float*)alloc((size_t)BT_ * D_ * 4);
  float* ycur = (float*)alloc((size_t)BT_ * D_ * 4);

  (void)hipMemcpyAsync(xcur, x, (size_t)BT_ * D_ * 4,
                       hipMemcpyDeviceToDevice, stream);
  (void)hipMemcpyAsync(ycur, y, (size_t)BT_ * D_ * 4,
                       hipMemcpyDeviceToDevice, stream);

  auto cvt = [&](const float* s, bf16* d, int n) {
    cvt_bf16_kernel<<<(n + 255) / 256, 256, 0, stream>>>(s, d, n);
  };
  cvt(Wq,  Wqb,  L_ * D_ * D_);
  cvt(Wk,  Wkb,  L_ * D_ * D_);
  cvt(Wv,  Wvb,  L_ * D_ * D_);
  cvt(Wox, Woxb, L_ * D_ * D_);
  cvt(Woy, Woyb, L_ * D_ * D_);
  cvt(fxw1, w1xb, L_ * D_ * F_);
  cvt(fxw2, w2xb, L_ * F_ * D_);
  cvt(fyw1, w1yb, L_ * D_ * F_);
  cvt(fyw2, w2yb, L_ * F_ * D_);

  auto gemm = [&](const bf16* A, const bf16* W, const float* bias,
                  const float* resid, float* outF, bf16* outB,
                  int M, int K, int N, int dg) {
    dim3 grid(M / 64, N / 128);
    gemm_bf16_wmma<<<grid, 256, 0, stream>>>(A, W, bias, resid, outF, outB,
                                             M, K, N, dg);
  };
  auto ln = [&](const float* X, const float* g, const float* b, bf16* out) {
    ln_kernel<<<BT_, 128, 0, stream>>>(X, g, b, out);
  };

  for (int layer = 0; layer < L_; ++layer) {
    const bf16* wq  = Wqb  + (size_t)layer * D_ * D_;
    const bf16* wk  = Wkb  + (size_t)layer * D_ * D_;
    const bf16* wv  = Wvb  + (size_t)layer * D_ * D_;
    const bf16* wox = Woxb + (size_t)layer * D_ * D_;
    const bf16* woy = Woyb + (size_t)layer * D_ * D_;
    const bf16* w1x = w1xb + (size_t)layer * D_ * F_;
    const bf16* w2x = w2xb + (size_t)layer * F_ * D_;
    const bf16* w1y = w1yb + (size_t)layer * D_ * F_;
    const bf16* w2y = w2yb + (size_t)layer * F_ * D_;

    // LN1
    ln(xcur, ln1xg + layer * D_, ln1xb + layer * D_, xn);
    ln(ycur, ln1yg + layer * D_, ln1yb + layer * D_, yn);

    // QKV: x-branch Q from yn, K/V from xn; y-branch Q from xn, K/V from yn
    gemm(yn, wq, nullptr, nullptr, nullptr, qxb, BT_, D_, D_, 0);
    gemm(xn, wk, nullptr, nullptr, nullptr, kxb, BT_, D_, D_, 0);
    gemm(xn, wv, nullptr, nullptr, nullptr, vxb, BT_, D_, D_, 0);
    gemm(xn, wq, nullptr, nullptr, nullptr, qyb, BT_, D_, D_, 0);
    gemm(yn, wk, nullptr, nullptr, nullptr, kyb, BT_, D_, D_, 0);
    gemm(yn, wv, nullptr, nullptr, nullptr, vyb, BT_, D_, D_, 0);

    // shared-softmax cross attention (flash streaming)
    attn_wmma<<<dim3(T_ / 16, H_, B_), 32, 0, stream>>>(
        qxb, kxb, vxb, qyb, kyb, vyb, oxb, oyb);

    // output projections + residual
    gemm(oxb, wox, box + layer * D_, xcur, xcur, nullptr, BT_, D_, D_, 0);
    gemm(oyb, woy, boy + layer * D_, ycur, ycur, nullptr, BT_, D_, D_, 0);

    // LN2 + FFN (exact GELU)
    ln(xcur, ln2xg + layer * D_, ln2xb + layer * D_, xn);
    ln(ycur, ln2yg + layer * D_, ln2yb + layer * D_, yn);
    gemm(xn, w1x, fxb1 + layer * F_, nullptr, nullptr, hx, BT_, D_, F_, 1);
    gemm(hx, w2x, fxb2 + layer * D_, xcur, xcur, nullptr, BT_, F_, D_, 0);
    gemm(yn, w1y, fyb1 + layer * F_, nullptr, nullptr, hy, BT_, D_, F_, 1);
    gemm(hy, w2y, fyb2 + layer * D_, ycur, ycur, nullptr, BT_, F_, D_, 0);
  }

  float* outp = (float*)d_out;
  (void)hipMemcpyAsync(outp, xcur, (size_t)BT_ * D_ * 4,
                       hipMemcpyDeviceToDevice, stream);
  (void)hipMemcpyAsync(outp + (size_t)BT_ * D_, ycur, (size_t)BT_ * D_ * 4,
                       hipMemcpyDeviceToDevice, stream);
}
